// CondFlow_84705345011867
// MI455X (gfx1250) — matrix operand order repaired and checked by reading, not compiled, not run on hardware
//
#include <hip/hip_runtime.h>
#include <hip/hip_bf16.h>

typedef __attribute__((ext_vector_type(16))) __bf16 v16bf;
typedef __attribute__((ext_vector_type(8)))  float  v8f;

#define NBINS 32
#define HDIM 256
#define DDIM 64
#define PRAW 97
#define PPAD 112
#define NLAYERS 4
#define ROWS_PER_BLOCK 128

// workspace layout in u32 units (prep kernel output: packed-bf16 B fragments,
// each 32x16 tile = 32 lanes * 8 dwords = 256 u32)
#define W1_OFF  0        // 16 ntiles * 2 ktiles
#define W1_NT   16
#define W1_KT   2
#define W2_OFF  8192     // 16 ntiles * 8 ktiles
#define W2_NT   16
#define W2_KT   8
#define SPL_OFF 40960    // 4 layers * 7 ntiles * 8 ktiles
#define SPL_NT  7
#define SPL_KT  8
#define AFF_OFF 98304    // 1 ntile * 8 ktiles
#define AFF_KT  8
#define TOT_TILES 392    // 32 + 128 + 224 + 8

__device__ __forceinline__ unsigned f2bf1(float f) {
  unsigned u = __float_as_uint(f);
  unsigned r = u + 0x7FFFu + ((u >> 16) & 1u);   // round-to-nearest-even
  return r >> 16;
}
__device__ __forceinline__ unsigned packbf(float a, float b) {
  return f2bf1(a) | (f2bf1(b) << 16);
}

union BF16x16 { unsigned u[8]; v16bf v; };

__device__ __forceinline__ v8f wmma_bf16(const unsigned* a, const unsigned* b, v8f c) {
  BF16x16 A, Bm;
#pragma unroll
  for (int i = 0; i < 8; ++i) { A.u[i] = a[i]; Bm.u[i] = b[i]; }
  return __builtin_amdgcn_wmma_f32_16x16x32_bf16(false, A.v, false, Bm.v,
                                                 (short)0, c, false, false);
}

__device__ __forceinline__ float softplusf(float u) {
  return (u > 20.f) ? u : log1pf(expf(u));
}

// ---------------------------------------------------------------------------
// Prep: repack all f32 weights into bf16 WMMA B-fragments in d_ws.
// B-fragment layout (32x16 tile): lane L holds col n = n0+(L&15),
// kb = (L>>4)*8, dword v holds K pair (kb+2(v&3)+16(v>>2), +1).
// ---------------------------------------------------------------------------
__global__ void cf_prep_weights(const float* __restrict__ w1,
                                const float* __restrict__ w2,
                                const float* __restrict__ spw,
                                const float* __restrict__ affw,
                                unsigned* __restrict__ ws) {
  int tid = blockIdx.x * blockDim.x + threadIdx.x;
  if (tid >= TOT_TILES * 32) return;
  int tile = tid >> 5, lane = tid & 31;

  const float* base; int ld, nlim, n0, k0;
  if (tile < 32) {                     // W1: [64,256]
    int nt = tile >> 1, kt = tile & 1;
    base = w1; ld = HDIM; nlim = HDIM; n0 = nt * 16; k0 = kt * 32;
  } else if (tile < 160) {             // W2: [256,256]
    int t = tile - 32, nt = t >> 3, kt = t & 7;
    base = w2; ld = HDIM; nlim = HDIM; n0 = nt * 16; k0 = kt * 32;
  } else if (tile < 384) {             // spline_w: [4,256,97] (pad N to 112)
    int t = tile - 160, layer = t / 56, r = t % 56, nt = r >> 3, kt = r & 7;
    base = spw + layer * HDIM * PRAW; ld = PRAW; nlim = PRAW;
    n0 = nt * 16; k0 = kt * 32;
  } else {                             // aff_w: [256,4] (pad N to 16)
    int kt = tile - 384;
    base = affw; ld = 4; nlim = 4; n0 = 0; k0 = kt * 32;
  }

  unsigned* dst = ws + tile * 256 + lane * 8;
  int n  = n0 + (lane & 15);
  int kh = k0 + ((lane >> 4) << 3);
  bool ok = (n < nlim);
#pragma unroll
  for (int v = 0; v < 8; ++v) {
    int k = kh + 2 * (v & 3) + ((v >> 2) << 4);
    float f0 = ok ? base[k * ld + n]       : 0.f;
    float f1 = ok ? base[(k + 1) * ld + n] : 0.f;
    dst[v] = packbf(f0, f1);
  }
}

// ---------------------------------------------------------------------------
// Main fused kernel: 256 threads = 8 waves, 128 rows/block, 16 rows/wave.
// Double-buffered LDS weight staging: next chunk's global loads are issued
// before the WMMA loop, stored to the alternate buffer after compute.
// ---------------------------------------------------------------------------
__global__ __launch_bounds__(256) void cf_main(
    const float* __restrict__ cond, const float* __restrict__ t,
    const float* __restrict__ b1,   const float* __restrict__ b2,
    const float* __restrict__ splb, const float* __restrict__ affb,
    const unsigned* __restrict__ wfrag, float* __restrict__ out) {

  __shared__ unsigned shm[36864];                 // 144 KB (<320 KB WGP LDS)
  unsigned* lat1 = shm;                           // 128 rows * 128 u32 (bf16x2)
  unsigned* lat2 = shm + 16384;
  unsigned* wbuf = shm + 32768;                   // 2 x 2048 u32 weight stage
  unsigned short* lat1h = (unsigned short*)lat1;
  unsigned short* lat2h = (unsigned short*)lat2;
  float* pbuf  = (float*)shm;                     // reuses lat1: 128*112 f32
  float* sabuf = (float*)shm + 14336;             // 128*4 f32 (after pbuf)

  const int tidb  = threadIdx.x;
  const int wave  = tidb >> 5, lane = tidb & 31;
  const int mbase = wave * 16;                    // block-local row base
  const int grow0 = blockIdx.x * ROWS_PER_BLOCK;
  const int nlane = lane & 15, hlf = lane >> 4;
  const int mtop  = hlf * 8;

  // ---- A fragments for GEMM1 straight from global cond (read once) ----
  unsigned a1[2][8];
  {
    int row = grow0 + mbase + nlane;
    const float* cp = cond + row * DDIM + hlf * 8;
#pragma unroll
    for (int kt = 0; kt < 2; ++kt) {
      const float* q = cp + kt * 32;
#pragma unroll
      for (int v = 0; v < 4; ++v) {
        a1[kt][v]     = packbf(q[2 * v],      q[2 * v + 1]);
        a1[kt][v + 4] = packbf(q[16 + 2 * v], q[16 + 2 * v + 1]);
      }
    }
  }

  // ---- GEMM1: latent1 = relu(cond @ W1 + b1), chunk = 512 u32 ----
  {
    const unsigned* srcb = wfrag + W1_OFF;
    if (tidb < 128) ((uint4*)wbuf)[tidb] = ((const uint4*)srcb)[tidb];
    __syncthreads();
    for (int j = 0; j < W1_NT; ++j) {
      const unsigned* cur = wbuf + (j & 1) * 2048;
      unsigned* nxt = wbuf + ((j + 1) & 1) * 2048;
      uint4 pf;
      bool have = (j + 1 < W1_NT) && (tidb < 128);
      if (have) pf = ((const uint4*)(srcb + (j + 1) * (W1_KT * 256)))[tidb];
      v8f acc = {};
#pragma unroll
      for (int kt = 0; kt < W1_KT; ++kt)
        acc = wmma_bf16(a1[kt], cur + kt * 256 + lane * 8, acc);
      int n = j * 16 + nlane;
      float bias = b1[n];
#pragma unroll
      for (int r = 0; r < 8; ++r) {
        float v = fmaxf(acc[r] + bias, 0.f);
        lat1h[(mbase + mtop + r) * HDIM + n] = (unsigned short)f2bf1(v);
      }
      if (have) ((uint4*)nxt)[tidb] = pf;
      __syncthreads();
    }
  }

  // ---- GEMM2: latent2 = relu(latent1 @ W2 + b2), chunk = 2048 u32 ----
  {
    const unsigned* srcb = wfrag + W2_OFF;
    ((uint4*)wbuf)[tidb]       = ((const uint4*)srcb)[tidb];
    ((uint4*)wbuf)[tidb + 256] = ((const uint4*)srcb)[tidb + 256];
    __syncthreads();
    for (int j = 0; j < W2_NT; ++j) {
      const unsigned* cur = wbuf + (j & 1) * 2048;
      unsigned* nxt = wbuf + ((j + 1) & 1) * 2048;
      uint4 pf0, pf1;
      bool have = (j + 1 < W2_NT);
      if (have) {
        const uint4* s4 = (const uint4*)(srcb + (j + 1) * (W2_KT * 256));
        pf0 = s4[tidb]; pf1 = s4[tidb + 256];
      }
      v8f acc = {};
      unsigned afr[8];
#pragma unroll
      for (int kt = 0; kt < W2_KT; ++kt) {
        const unsigned* p = lat1 + (mbase + nlane) * 128 + kt * 16 + hlf * 4;
#pragma unroll
        for (int v = 0; v < 4; ++v) { afr[v] = p[v]; afr[v + 4] = p[v + 8]; }
        acc = wmma_bf16(afr, cur + kt * 256 + lane * 8, acc);
      }
      int n = j * 16 + nlane;
      float bias = b2[n];
#pragma unroll
      for (int r = 0; r < 8; ++r) {
        float v = fmaxf(acc[r] + bias, 0.f);
        lat2h[(mbase + mtop + r) * HDIM + n] = (unsigned short)f2bf1(v);
      }
      if (have) { ((uint4*)nxt)[tidb] = pf0; ((uint4*)nxt)[tidb + 256] = pf1; }
      __syncthreads();
    }
  }

  // ---- Affine head: sa = latent2 @ aff_w + aff_b  (N padded to 16) ----
  {
    const unsigned* srcb = wfrag + AFF_OFF;
    ((uint4*)wbuf)[tidb]       = ((const uint4*)srcb)[tidb];
    ((uint4*)wbuf)[tidb + 256] = ((const uint4*)srcb)[tidb + 256];
    __syncthreads();
    v8f acc = {};
    unsigned afr[8];
#pragma unroll
    for (int kt = 0; kt < AFF_KT; ++kt) {
      const unsigned* p = lat2 + (mbase + nlane) * 128 + kt * 16 + hlf * 4;
#pragma unroll
      for (int v = 0; v < 4; ++v) { afr[v] = p[v]; afr[v + 4] = p[v + 8]; }
      acc = wmma_bf16(afr, wbuf + kt * 256 + lane * 8, acc);
    }
    if (nlane < 4) {
      float bias = affb[nlane];
#pragma unroll
      for (int r = 0; r < 8; ++r)
        sabuf[(mbase + mtop + r) * 4 + nlane] = acc[r] + bias;
    }
  }
  __syncthreads();

  // ---- x = t * exp(sa1) + sa0 (lanes 0..15 each own one row) ----
  float x = 0.f;
  if (lane < 16) {
    int blr = mbase + lane;
    float sa0 = sabuf[blr * 4 + 0], sa1 = sabuf[blr * 4 + 1];
    x = t[grow0 + blr] * expf(sa1) + sa0;
  }

  // ---- 4 spline layers (applied last-first): p = latent2@spline_w[i]+b ----
  for (int layer = NLAYERS - 1; layer >= 0; --layer) {
    const unsigned* srcb = wfrag + SPL_OFF + (layer * SPL_NT * SPL_KT) * 256;
    ((uint4*)wbuf)[tidb]       = ((const uint4*)srcb)[tidb];
    ((uint4*)wbuf)[tidb + 256] = ((const uint4*)srcb)[tidb + 256];
    __syncthreads();
    for (int j = 0; j < SPL_NT; ++j) {
      const unsigned* cur = wbuf + (j & 1) * 2048;
      unsigned* nxt = wbuf + ((j + 1) & 1) * 2048;
      uint4 pf0, pf1;
      bool have = (j + 1 < SPL_NT);
      if (have) {
        const uint4* s4 = (const uint4*)(srcb + (j + 1) * (SPL_KT * 256));
        pf0 = s4[tidb]; pf1 = s4[tidb + 256];
      }
      v8f acc = {};
      unsigned afr[8];
#pragma unroll
      for (int kt = 0; kt < SPL_KT; ++kt) {
        const unsigned* p = lat2 + (mbase + nlane) * 128 + kt * 16 + hlf * 4;
#pragma unroll
        for (int v = 0; v < 4; ++v) { afr[v] = p[v]; afr[v + 4] = p[v + 8]; }
        acc = wmma_bf16(afr, cur + kt * 256 + lane * 8, acc);
      }
      int n = j * 16 + nlane;
      float bias = (n < PRAW) ? splb[layer * PRAW + n] : 0.f;
#pragma unroll
      for (int r = 0; r < 8; ++r)
        pbuf[(mbase + mtop + r) * PPAD + n] = acc[r] + bias;
      if (have) { ((uint4*)nxt)[tidb] = pf0; ((uint4*)nxt)[tidb + 256] = pf1; }
      __syncthreads();
    }

    // rational-quadratic spline forward, one row per lane (f32)
    if (lane < 16) {
      int blr = mbase + lane;
      const float* pr = pbuf + blr * PPAD;
      float mw = -1e30f, mh = -1e30f;
      for (int k = 0; k < NBINS; ++k) {
        mw = fmaxf(mw, pr[k]);
        mh = fmaxf(mh, pr[NBINS + k]);
      }
      float sw = 0.f, sh = 0.f;
      for (int k = 0; k < NBINS; ++k) {
        sw += expf(pr[k] - mw);
        sh += expf(pr[NBINS + k] - mh);
      }
      const float TOT = 6.0f - NBINS * 1e-5f;
      float cwm = TOT / sw, chm = TOT / sh;
      float cx = -3.f, cy = -3.f, xk = -3.f, yk = -3.f, wk = 1.f, hk = 1.f;
      int idx = 0;
      for (int k = 0; k < NBINS; ++k) {
        float w = expf(pr[k] - mw) * cwm + 1e-5f;
        float h = expf(pr[NBINS + k] - mh) * chm + 1e-5f;
        if (k == 0) { wk = w; hk = h; }
        else if (x >= cx) { idx = k; xk = cx; yk = cy; wk = w; hk = h; }
        cx += w; cy += h;
      }
      // knot slopes: softplus(u + log(exp(1-1e-5)-1)) + 1e-5
      float dk  = softplusf(pr[2 * NBINS + idx]     + 0.5413090f) + 1e-5f;
      float dk1 = softplusf(pr[2 * NBINS + idx + 1] + 0.5413090f) + 1e-5f;
      float z  = fminf(fmaxf((x - xk) / wk, 0.f), 1.f);
      float s  = hk / wk;
      float z1 = z * (1.f - z);
      float num = hk * (s * z * z + dk * z1);
      float den = s + (dk1 + dk - 2.f * s) * z1;
      float y = yk + num / den;
      x = (x < -3.f || x > 3.f) ? x : y;
    }
    __syncthreads();
  }

  // ---- f2 + normal CDF ----
  if (lane < 16) {
    int blr = mbase + lane;
    float sa2 = sabuf[blr * 4 + 2], sa3 = sabuf[blr * 4 + 3];
    float xf = x * expf(sa3) + sa2;
    out[grow0 + blr] = 0.5f * erfcf(-xf * 0.70710678118654752f);
  }
}

extern "C" void kernel_launch(void* const* d_in, const int* in_sizes, int n_in,
                              void* d_out, int out_size, void* d_ws, size_t ws_size,
                              hipStream_t stream) {
  const float* cond = (const float*)d_in[0];
  const float* t    = (const float*)d_in[1];
  const float* w1   = (const float*)d_in[2];
  const float* b1   = (const float*)d_in[3];
  const float* w2   = (const float*)d_in[4];
  const float* b2   = (const float*)d_in[5];
  const float* spw  = (const float*)d_in[6];
  const float* splb = (const float*)d_in[7];
  const float* affw = (const float*)d_in[8];
  const float* affb = (const float*)d_in[9];
  float* out   = (float*)d_out;
  unsigned* ws = (unsigned*)d_ws;

  int B = in_sizes[1];  // t has shape [B,1]

  cf_prep_weights<<<(TOT_TILES * 32 + 255) / 256, 256, 0, stream>>>(w1, w2, spw, affw, ws);
  cf_main<<<B / ROWS_PER_BLOCK, 256, 0, stream>>>(cond, t, b1, b2, splb, affb, ws, out);
}
